// EncoderLayer_29343216566369
// MI455X (gfx1250) — matrix operand order repaired and checked by reading, not compiled, not run on hardware
//
#include <hip/hip_runtime.h>
#include <hip/hip_bf16.h>

typedef __attribute__((ext_vector_type(16))) _Float16 v16h;
typedef __attribute__((ext_vector_type(8)))  float    v8f;

#define SEQ_L   128
#define NBAT    4096
#define EMB     12
#define HID     32
#define PH_STR  136   // P row stride in halves: 272B rows, 16B-aligned chunks

union ABOp { v16h h; uint4 u[2]; };

__device__ __forceinline__ float half_sum(float v) {
  #pragma unroll
  for (int off = 8; off >= 1; off >>= 1) v += __shfl_xor(v, off, 32);
  return v;
}
__device__ __forceinline__ float half_max(float v) {
  #pragma unroll
  for (int off = 8; off >= 1; off >>= 1) v = fmaxf(v, __shfl_xor(v, off, 32));
  return v;
}
__device__ __forceinline__ float fast_tanh(float x) {
  float e = __expf(2.f * x);                 // v_exp_f32-based, branch free
  return 1.f - __fdividef(2.f, e + 1.f);     // tanh(0)=0, saturates to +-1
}

__global__ __launch_bounds__(128)
void encoder_layer_kernel(const float* __restrict__ x,
    const float* __restrict__ in_proj_w, const float* __restrict__ in_proj_b,
    const float* __restrict__ out_w,     const float* __restrict__ out_b,
    const float* __restrict__ w1, const float* __restrict__ b1,
    const float* __restrict__ w2, const float* __restrict__ b2,
    const float* __restrict__ w3, const float* __restrict__ b3,
    const float* __restrict__ g1, const float* __restrict__ be1,
    const float* __restrict__ g2, const float* __restrict__ be2,
    float* __restrict__ out)
{
  __shared__ __align__(16) float    xs[SEQ_L][16];        // x copy, cols 12..15 = 0
  __shared__ __align__(16) _Float16 qh[SEQ_L][32];        // Q/sqrt(E); reused as `outs` f32[128][16]
  __shared__ __align__(16) _Float16 kh[SEQ_L][32];        // K, cols 12..31 = 0
  __shared__ __align__(16) _Float16 vt[16][SEQ_L];        // V transposed, rows 12..15 = 0
  __shared__ __align__(16) _Float16 ph[SEQ_L * PH_STR];   // P; reused as zh/h1h/h2h/zs
  __shared__ __align__(16) _Float16 ah[SEQ_L * 32];       // attn f16, cols 12..31 = 0
  // weights staged once per block (f16, pre-transposed to B-operand layout)
  __shared__ __align__(16) _Float16 owhT[16 * 32];        // owhT[e][k] = out_w[e,k], pads 0
  __shared__ __align__(16) _Float16 w1h[32 * 32];         // w1h[g][k]  = w1[g,k], k>=12 -> 0
  __shared__ __align__(16) _Float16 w2h[32 * 32];         // w2h[g][k]  = w2[g,k]
  __shared__ __align__(16) _Float16 w3h[16 * 32];         // w3h[e][k]  = w3[e,k], e>=12 -> 0
  __shared__ float wiS[36 * 12];                          // in_proj_w (f32)
  __shared__ float wiB[36];
  __shared__ float c_outb[12], c_b1[32], c_b2[32], c_b3[12];
  __shared__ float c_g1[12], c_be1[12], c_g2[12], c_be2[12];

  const int n    = blockIdx.x;
  const int t    = threadIdx.x;
  const int lane = t & 31;
  const int wave = t >> 5;

  // ---------------- phase 0: stage x row + all weights ----------------
  {
    const float* xr = x + ((size_t)t * NBAT + n) * EMB;
    #pragma unroll
    for (int e = 0; e < EMB; ++e) xs[t][e] = xr[e];
    #pragma unroll
    for (int e = EMB; e < 16; ++e) xs[t][e] = 0.f;
  }
  for (int i = t; i < 36 * EMB; i += 128) wiS[i] = in_proj_w[i];
  for (int i = t; i < 36;       i += 128) wiB[i] = in_proj_b[i];
  for (int i = t; i < 16 * 32; i += 128) {
    int e = i >> 5, k = i & 31;
    owhT[i] = (e < EMB && k < EMB) ? (_Float16)out_w[e * EMB + k] : (_Float16)0.f;
    w3h[i]  = (e < EMB)            ? (_Float16)w3[e * HID + k]    : (_Float16)0.f;
  }
  for (int i = t; i < 32 * 32; i += 128) {
    int g = i >> 5, k = i & 31;
    w1h[i] = (k < EMB) ? (_Float16)w1[g * EMB + k] : (_Float16)0.f;
    w2h[i] = (_Float16)w2[i];
  }
  if (t < EMB) {
    c_outb[t] = out_b[t]; c_b3[t] = b3[t];
    c_g1[t] = g1[t]; c_be1[t] = be1[t]; c_g2[t] = g2[t]; c_be2[t] = be2[t];
  }
  if (t < HID) { c_b1[t] = b1[t]; c_b2[t] = b2[t]; }
  __syncthreads();

  // ---------------- QKV projection (per-thread row, LDS weights) ----------------
  {
    float xr[EMB];
    #pragma unroll
    for (int e = 0; e < EMB; ++e) xr[e] = xs[t][e];
    const float scale = 0.28867513459481287f;  // 1/sqrt(12)
    #pragma unroll
    for (int f = 0; f < 3 * EMB; ++f) {
      float acc = wiB[f];
      #pragma unroll
      for (int e = 0; e < EMB; ++e) acc += xr[e] * wiS[f * EMB + e];
      if      (f < EMB)     qh[t][f]           = (_Float16)(acc * scale);
      else if (f < 2 * EMB) kh[t][f - EMB]     = (_Float16)acc;
      else                  vt[f - 2 * EMB][t] = (_Float16)acc;
    }
    #pragma unroll
    for (int e = EMB; e < 32; ++e) { qh[t][e] = (_Float16)0.f; kh[t][e] = (_Float16)0.f; }
    #pragma unroll
    for (int e = EMB; e < 16; ++e) vt[e][t] = (_Float16)0.f;
    #pragma unroll
    for (int e = 16; e < 32; ++e) ah[t * 32 + e] = (_Float16)0.f;  // pad for out-proj A
  }
  __syncthreads();

  // operand geometry (selects on offsets only, never on loads)
  const int kbA   = (lane < 16) ? 0 : 8;    // A: lo lanes K{0..7,16..23}, hi {8..15,24..31}
  const int kofsB = (lane < 16) ? 0 : 8;    // B: lo lanes K{0..7,16..23}, hi {8..15,24..31}
  const int nn    = lane & 15;
  const int rhalf = (lane < 16) ? 0 : 8;    // C: lo lanes rows r, hi lanes rows r+8
  const int idx12 = (nn < EMB) ? nn : 0;

  // ---------------- scores = QK^T + softmax (wave owns rows [wave*32,+32)) ----------------
  #pragma unroll
  for (int mi = 0; mi < 2; ++mi) {
    const int mt   = wave * 2 + mi;
    const int mrow = mt * 16 + nn;
    ABOp A;
    A.u[0] = *reinterpret_cast<const uint4*>(&qh[mrow][kbA]);
    A.u[1] = *reinterpret_cast<const uint4*>(&qh[mrow][16 + kbA]);

    v8f c[8];
    #pragma unroll
    for (int j = 0; j < 8; ++j) {
      ABOp B;
      B.u[0] = *reinterpret_cast<const uint4*>(&kh[j * 16 + nn][kofsB]);
      B.u[1] = *reinterpret_cast<const uint4*>(&kh[j * 16 + nn][kofsB + 16]);
      v8f z = {};
      c[j] = __builtin_amdgcn_wmma_f32_16x16x32_f16(false, A.h, false, B.h,
                                                    (short)0, z, false, false);
    }
    float rmax[8], rsum[8];
    #pragma unroll
    for (int r = 0; r < 8; ++r) {
      float m0 = c[0][r];
      #pragma unroll
      for (int j = 1; j < 8; ++j) m0 = fmaxf(m0, c[j][r]);
      rmax[r] = half_max(m0);
      rsum[r] = 0.f;
    }
    #pragma unroll
    for (int j = 0; j < 8; ++j)
      #pragma unroll
      for (int r = 0; r < 8; ++r) {
        float e = __expf(c[j][r] - rmax[r]);
        c[j][r] = e; rsum[r] += e;
      }
    #pragma unroll
    for (int r = 0; r < 8; ++r) rsum[r] = __fdividef(1.f, half_sum(rsum[r]));
    const int rb = mt * 16 + rhalf;
    #pragma unroll
    for (int r = 0; r < 8; ++r)
      #pragma unroll
      for (int j = 0; j < 8; ++j)
        ph[(rb + r) * PH_STR + j * 16 + nn] = (_Float16)(c[j][r] * rsum[r]);
  }

  // ---------------- attn = P @ V (4 chained K=32 WMMAs per M-tile) ----------------
  #pragma unroll
  for (int mi = 0; mi < 2; ++mi) {
    const int mt = wave * 2 + mi;
    const int m  = mt * 16 + nn;
    v8f acc = {};
    #pragma unroll
    for (int kc = 0; kc < 4; ++kc) {
      const int base = kc * 32;
      ABOp A, B;
      A.u[0] = *reinterpret_cast<const uint4*>(&ph[m * PH_STR + base + kbA]);
      A.u[1] = *reinterpret_cast<const uint4*>(&ph[m * PH_STR + base + 16 + kbA]);
      B.u[0] = *reinterpret_cast<const uint4*>(&vt[nn][base + kofsB]);
      B.u[1] = *reinterpret_cast<const uint4*>(&vt[nn][base + kofsB + 16]);
      acc = __builtin_amdgcn_wmma_f32_16x16x32_f16(false, A.h, false, B.h,
                                                   (short)0, acc, false, false);
    }
    const int rb = mt * 16 + rhalf;
    #pragma unroll
    for (int r = 0; r < 8; ++r) ah[(rb + r) * 32 + nn] = (_Float16)acc[r]; // cols 12..15 = 0
  }
  __syncthreads();   // all waves done with qh/kh/vt/ph -> safe to re-use that LDS

  // LDS reuse (all row-ranges are wave-private from here on)
  _Float16* zh   = ph;                                      // [128][32] LN1 out, f16
  _Float16* h1h  = ph + 4096;                               // [128][32]
  _Float16* h2h  = ph + 8192;                               // [128][32]
  float*    zs   = reinterpret_cast<float*>(ph + 12288);    // [128][16] LN1 out, f32
  float*    outs = reinterpret_cast<float*>(&qh[0][0]);     // [128][16] final result

  // ---------------- out-proj + residual + LN1 (WMMA, C-layout) ----------------
  #pragma unroll
  for (int mi = 0; mi < 2; ++mi) {
    const int mt = wave * 2 + mi;
    const int rb = mt * 16 + rhalf;
    ABOp A, B;
    A.u[0] = *reinterpret_cast<const uint4*>(&ah[(mt * 16 + nn) * 32 + kbA]);
    A.u[1] = *reinterpret_cast<const uint4*>(&ah[(mt * 16 + nn) * 32 + 16 + kbA]);
    B.u[0] = *reinterpret_cast<const uint4*>(&owhT[nn * 32 + kofsB]);
    B.u[1] = *reinterpret_cast<const uint4*>(&owhT[nn * 32 + kofsB + 16]);
    const float bias = (nn < EMB) ? c_outb[idx12] : 0.f;
    v8f c;
    #pragma unroll
    for (int r = 0; r < 8; ++r) c[r] = bias;
    c = __builtin_amdgcn_wmma_f32_16x16x32_f16(false, A.h, false, B.h,
                                               (short)0, c, false, false);
    #pragma unroll
    for (int r = 0; r < 8; ++r) {
      float y  = c[r] + xs[rb + r][nn];          // cols >= 12 are 0 + 0
      float mu = half_sum(y) * (1.f / EMB);
      float d  = (nn < EMB) ? (y - mu) : 0.f;
      float iv = rsqrtf(half_sum(d * d) * (1.f / EMB) + 1e-5f);
      float zf = (nn < EMB) ? ((y - mu) * iv * c_g1[idx12] + c_be1[idx12]) : 0.f;
      zh[(rb + r) * 32 + nn] = (_Float16)zf;
      zh[(rb + r) * 32 + 16 + nn] = (_Float16)0.f;   // zero K-pad 16..31
      zs[(rb + r) * 16 + nn] = zf;
    }
  }

  // ---------------- MLP layer1: h1 = tanh(z @ W1^T + b1)  (K=12->32) ----------------
  #pragma unroll
  for (int mi = 0; mi < 2; ++mi) {
    const int mt = wave * 2 + mi;
    const int rb = mt * 16 + rhalf;
    ABOp A;
    A.u[0] = *reinterpret_cast<const uint4*>(&zh[(mt * 16 + nn) * 32 + kbA]);
    A.u[1] = *reinterpret_cast<const uint4*>(&zh[(mt * 16 + nn) * 32 + 16 + kbA]);
    #pragma unroll
    for (int nt = 0; nt < 2; ++nt) {
      const int g = nt * 16 + nn;
      ABOp B;
      B.u[0] = *reinterpret_cast<const uint4*>(&w1h[g * 32 + kofsB]);
      B.u[1] = *reinterpret_cast<const uint4*>(&w1h[g * 32 + kofsB + 16]);
      v8f c;
      #pragma unroll
      for (int r = 0; r < 8; ++r) c[r] = c_b1[g];
      c = __builtin_amdgcn_wmma_f32_16x16x32_f16(false, A.h, false, B.h,
                                                 (short)0, c, false, false);
      #pragma unroll
      for (int r = 0; r < 8; ++r)
        h1h[(rb + r) * 32 + g] = (_Float16)fast_tanh(c[r]);
    }
  }

  // ---------------- MLP layer2: h2 = tanh(h1 @ W2^T + b2)  (K=32 exact) ----------------
  #pragma unroll
  for (int mi = 0; mi < 2; ++mi) {
    const int mt = wave * 2 + mi;
    const int rb = mt * 16 + rhalf;
    ABOp A;
    A.u[0] = *reinterpret_cast<const uint4*>(&h1h[(mt * 16 + nn) * 32 + kbA]);
    A.u[1] = *reinterpret_cast<const uint4*>(&h1h[(mt * 16 + nn) * 32 + 16 + kbA]);
    #pragma unroll
    for (int nt = 0; nt < 2; ++nt) {
      const int g = nt * 16 + nn;
      ABOp B;
      B.u[0] = *reinterpret_cast<const uint4*>(&w2h[g * 32 + kofsB]);
      B.u[1] = *reinterpret_cast<const uint4*>(&w2h[g * 32 + kofsB + 16]);
      v8f c;
      #pragma unroll
      for (int r = 0; r < 8; ++r) c[r] = c_b2[g];
      c = __builtin_amdgcn_wmma_f32_16x16x32_f16(false, A.h, false, B.h,
                                                 (short)0, c, false, false);
      #pragma unroll
      for (int r = 0; r < 8; ++r)
        h2h[(rb + r) * 32 + g] = (_Float16)fast_tanh(c[r]);
    }
  }

  // ------- MLP layer3 + residual + LN2 + stage output (K=32, 12 valid cols) -------
  #pragma unroll
  for (int mi = 0; mi < 2; ++mi) {
    const int mt = wave * 2 + mi;
    const int rb = mt * 16 + rhalf;
    ABOp A, B;
    A.u[0] = *reinterpret_cast<const uint4*>(&h2h[(mt * 16 + nn) * 32 + kbA]);
    A.u[1] = *reinterpret_cast<const uint4*>(&h2h[(mt * 16 + nn) * 32 + 16 + kbA]);
    B.u[0] = *reinterpret_cast<const uint4*>(&w3h[nn * 32 + kofsB]);
    B.u[1] = *reinterpret_cast<const uint4*>(&w3h[nn * 32 + kofsB + 16]);
    const float bias = (nn < EMB) ? c_b3[idx12] : 0.f;
    v8f c;
    #pragma unroll
    for (int r = 0; r < 8; ++r) c[r] = bias;
    c = __builtin_amdgcn_wmma_f32_16x16x32_f16(false, A.h, false, B.h,
                                               (short)0, c, false, false);
    #pragma unroll
    for (int r = 0; r < 8; ++r) {
      float y  = fast_tanh(c[r]) + zs[(rb + r) * 16 + nn];  // cols >= 12: 0 + 0
      float mu = half_sum(y) * (1.f / EMB);
      float d  = (nn < EMB) ? (y - mu) : 0.f;
      float iv = rsqrtf(half_sum(d * d) * (1.f / EMB) + 1e-5f);
      outs[(rb + r) * 16 + nn] =
          (nn < EMB) ? ((y - mu) * iv * c_g2[idx12] + c_be2[idx12]) : 0.f;
    }
  }
  __syncthreads();

  // ---------------- coalesced per-thread row store ----------------
  {
    float* orow = out + ((size_t)t * NBAT + n) * EMB;
    #pragma unroll
    for (int e = 0; e < EMB; ++e) orow[e] = outs[t * 16 + e];
  }
}

extern "C" void kernel_launch(void* const* d_in, const int* in_sizes, int n_in,
                              void* d_out, int out_size, void* d_ws, size_t ws_size,
                              hipStream_t stream) {
  const float* x         = (const float*)d_in[0];
  const float* in_proj_w = (const float*)d_in[1];
  const float* in_proj_b = (const float*)d_in[2];
  const float* out_w     = (const float*)d_in[3];
  const float* out_b     = (const float*)d_in[4];
  const float* w1        = (const float*)d_in[5];
  const float* b1        = (const float*)d_in[6];
  const float* w2        = (const float*)d_in[7];
  const float* b2        = (const float*)d_in[8];
  const float* w3        = (const float*)d_in[9];
  const float* b3        = (const float*)d_in[10];
  const float* g1        = (const float*)d_in[11];
  const float* be1       = (const float*)d_in[12];
  const float* g2        = (const float*)d_in[13];
  const float* be2       = (const float*)d_in[14];

  encoder_layer_kernel<<<dim3(NBAT), dim3(128), 0, stream>>>(
      x, in_proj_w, in_proj_b, out_w, out_b,
      w1, b1, w2, b2, w3, b3, g1, be1, g2, be2,
      (float*)d_out);
}
